// MemoryComponent_45389214384574
// MI455X (gfx1250) — matrix operand order repaired and checked by reading, not compiled, not run on hardware
//
#include <hip/hip_runtime.h>

// ---------------- problem constants ----------------
#define NT   16384   // B*S tokens
#define DDIM 2048    // hidden size
#define IDIM 4096    // mlp intermediate
#define NH   16      // heads
#define HDIM 128     // head dim
#define PVN  64      // persistent vectors

typedef __bf16 bf16;
typedef __attribute__((ext_vector_type(8)))  __bf16 v8bf;
typedef __attribute__((ext_vector_type(16))) __bf16 v16bf;
typedef __attribute__((ext_vector_type(8)))  float  v8f;

union Frag16 { v16bf v; v8bf h[2]; };

__device__ __forceinline__ bf16 f2bf(float f) {
  unsigned u = __builtin_bit_cast(unsigned, f);
  unsigned r = u + 0x7FFFu + ((u >> 16) & 1u);     // round-to-nearest-even
  unsigned short s = (unsigned short)(r >> 16);
  return __builtin_bit_cast(bf16, s);
}
__device__ __forceinline__ float bf2f(bf16 b) {
  unsigned short s = __builtin_bit_cast(unsigned short, b);
  unsigned u = ((unsigned)s) << 16;
  return __builtin_bit_cast(float, u);
}

// CDNA5 async global->LDS copy (16B per lane), tracked by ASYNCcnt.
// VDST = LDS byte address (low 32 bits of generic shared pointer, per ISA 10.2).
__device__ __forceinline__ void async_copy16(unsigned lds_off, const void* gptr) {
  asm volatile("global_load_async_to_lds_b128 %0, %1, off"
               :: "v"(lds_off), "v"(gptr)
               : "memory");
}

// ---------------- f32 -> bf16 conversion ----------------
__global__ __launch_bounds__(256) void f2bf_kernel(const float* __restrict__ in,
                                                   bf16* __restrict__ out, size_t n) {
  for (size_t i = (size_t)blockIdx.x * 256 + threadIdx.x; i < n; i += (size_t)gridDim.x * 256)
    out[i] = f2bf(in[i]);
}

// ---------------- LayerNorm (row per block) ----------------
__global__ __launch_bounds__(256) void ln_kernel(const float* __restrict__ x,
                                                 const float* __restrict__ g,
                                                 const float* __restrict__ b,
                                                 bf16* __restrict__ out) {
  __shared__ float red[256];
  const int row = blockIdx.x;
  const float* xr = x + (size_t)row * DDIM;
  float s = 0.f, s2 = 0.f;
  for (int i = threadIdx.x; i < DDIM; i += 256) { float v = xr[i]; s += v; s2 += v * v; }
  red[threadIdx.x] = s; __syncthreads();
  for (int o = 128; o > 0; o >>= 1) { if (threadIdx.x < o) red[threadIdx.x] += red[threadIdx.x + o]; __syncthreads(); }
  float mean = red[0] * (1.f / DDIM); __syncthreads();
  red[threadIdx.x] = s2; __syncthreads();
  for (int o = 128; o > 0; o >>= 1) { if (threadIdx.x < o) red[threadIdx.x] += red[threadIdx.x + o]; __syncthreads(); }
  float var = red[0] * (1.f / DDIM) - mean * mean;
  float rstd = rsqrtf(var + 1e-12f);
  bf16* orow = out + (size_t)row * DDIM;
  for (int i = threadIdx.x; i < DDIM; i += 256)
    orow[i] = f2bf((xr[i] - mean) * rstd * g[i] + b[i]);
}

// ---------------- shared GEMM pieces ----------------
#define BM 128
#define BN 128
#define BKK 32
#define SA 40
#define SBS 40

// Load fragments for one 32-deep K-step from LDS and run the 8 WMMAs of a
// 32x64 per-wave C sub-tile. Fragment addressing matches the CDNA5 wave32
// A (16x32) / B (32x16) bf16 VGPR layouts (ISA 7.12.2).
__device__ __forceinline__ void wmma_kstep(const bf16* __restrict__ Ab,
                                           const bf16* __restrict__ Bb,
                                           v8f acc[2][4], int wm, int wn,
                                           int l, int hf) {
  Frag16 af[2], bg[4];
  #pragma unroll
  for (int mt = 0; mt < 2; ++mt) {
    int row = wm * 32 + mt * 16 + l;
    af[mt].h[0] = *(const v8bf*)(Ab + row * SA + hf * 8);        // K 0..7 / 8..15
    af[mt].h[1] = *(const v8bf*)(Ab + row * SA + 16 + hf * 8);   // K 16..23 / 24..31
  }
  #pragma unroll
  for (int nt = 0; nt < 4; ++nt) {
    int col = wn * 64 + nt * 16 + l;
    bg[nt].h[0] = *(const v8bf*)(Bb + col * SBS + hf * 8);       // K 0..7 / 8..15
    bg[nt].h[1] = *(const v8bf*)(Bb + col * SBS + 16 + hf * 8);  // K 16..23 / 24..31
  }
  #pragma unroll
  for (int mt = 0; mt < 2; ++mt)
    #pragma unroll
    for (int nt = 0; nt < 4; ++nt)
      acc[mt][nt] = __builtin_amdgcn_wmma_f32_16x16x32_bf16(
          false, af[mt].v, false, bg[nt].v, (short)0, acc[mt][nt], false, false);
}

template<int MODE>
__device__ __forceinline__ void gemm_epilogue(
    v8f acc[2][4], int m0, int n0, int wm, int wn, int l, int hf,
    float* __restrict__ outF, bf16* __restrict__ outB, int ldc,
    const float* __restrict__ bias, const float* __restrict__ res,
    const float* __restrict__ aux1, const float* __restrict__ aux2,
    const bf16* __restrict__ maskp, float scale)
{
  #pragma unroll
  for (int mt = 0; mt < 2; ++mt) {
    #pragma unroll
    for (int nt = 0; nt < 4; ++nt) {
      int col = n0 + wn * 64 + nt * 16 + l;
      #pragma unroll
      for (int r = 0; r < 8; ++r) {
        int row = m0 + wm * 32 + mt * 16 + hf * 8 + r;
        size_t o = (size_t)row * ldc + col;
        float v = acc[mt][nt][r];
        if (MODE == 0) {                       // q = acc + bias            -> f32
          outF[o] = v + bias[col];
        } else if (MODE == 1) {                // pm = acc + bias + hidden  -> f32 + bf16
          float x = v + bias[col] + res[o]; outF[o] = x; outB[o] = f2bf(x);
        } else if (MODE == 2) {                // H = relu(acc + bias)      -> bf16
          float x = v + bias[col]; outB[o] = f2bf(x > 0.f ? x : 0.f);
        } else if (MODE == 3) {                // G = (acc+bias-res)*scale  -> bf16
          float x = (v + bias[col] - res[o]) * scale; outB[o] = f2bf(x);
        } else if (MODE == 4) {                // dHm = relu'(H) * acc      -> bf16
          outB[o] = f2bf(bf2f(maskp[o]) > 0.f ? v : 0.f);
        } else if (MODE == 5) {                // new_w = .99*w + .9*m - .01*acc -> bf16
          float x = 0.99f * aux1[o] + 0.9f * aux2[o] - 0.01f * v; outB[o] = f2bf(x);
        } else {                               // out = acc + bias + pm     -> f32
          outF[o] = v + bias[col] + res[o];
        }
      }
    }
  }
}

// ---------------- async double-buffered WMMA GEMM (A row-major [M][K], B [N][K]) ----------------
// gfx1250 GLOBAL_LOAD_ASYNC_TO_LDS_B128 + ASYNCcnt pipeline, LDS double buffer,
// last K-step peeled so the steady-state loop has no wait-selection branch.
template<int MODE>
__global__ __launch_bounds__(256) void gemm_bf16_async(
    const bf16* __restrict__ A, const bf16* __restrict__ B,
    int M, int N, int K, int lda, int ldb,
    float* __restrict__ outF, bf16* __restrict__ outB, int ldc,
    const float* __restrict__ bias, const float* __restrict__ res, float scale)
{
  __shared__ __align__(16) bf16 As[2][BM * SA];
  __shared__ __align__(16) bf16 Bs[2][BN * SBS];
  const int tid  = threadIdx.x;
  const int lane = tid & 31, wid = tid >> 5;
  const int wm = wid & 3, wn = wid >> 2;
  const int l = lane & 15, hf = lane >> 4;
  const int m0 = blockIdx.y * BM, n0 = blockIdx.x * BN;

  unsigned asb[2], bsb[2];
  asb[0] = (unsigned)(size_t)&As[0][0]; asb[1] = (unsigned)(size_t)&As[1][0];
  bsb[0] = (unsigned)(size_t)&Bs[0][0]; bsb[1] = (unsigned)(size_t)&Bs[1][0];

  auto issue = [&](int kt, int bb) {
    int k0 = kt * BKK;
    #pragma unroll
    for (int it = 0; it < 2; ++it) {
      int idx = tid + it * 256;
      int r = idx >> 2, q = idx & 3;              // 128 rows x 4 chunks of 8 bf16
      async_copy16(asb[bb] + (unsigned)((r * SA + q * 8) * 2),
                   (const void*)(A + (size_t)(m0 + r) * lda + k0 + q * 8));
    }
    #pragma unroll
    for (int it = 0; it < 2; ++it) {
      int idx = tid + it * 256;
      int r = idx >> 2, q = idx & 3;
      async_copy16(bsb[bb] + (unsigned)((r * SBS + q * 8) * 2),
                   (const void*)(B + (size_t)(n0 + r) * ldb + k0 + q * 8));
    }
  };

  v8f acc[2][4];
  #pragma unroll
  for (int i = 0; i < 2; ++i)
    #pragma unroll
    for (int j = 0; j < 4; ++j)
      #pragma unroll
      for (int e = 0; e < 8; ++e) acc[i][j][e] = 0.0f;

  const int nt = K / BKK;                         // >= 64 for all our shapes
  issue(0, 0);
  issue(1, 1);

  for (int t = 0; t < nt - 1; ++t) {
    asm volatile("s_wait_asynccnt 0x4" ::: "memory");  // tile t landed (t+1 in flight)
    __syncthreads();                                   // tile t visible to all waves
    const bf16* Ab = &As[t & 1][0];
    const bf16* Bb = &Bs[t & 1][0];
    // fragment reads must complete in every wave before buf[t&1] is overwritten
    // by tile t+2's async writes, so: ds reads -> dscnt flush -> barrier -> issue
    Frag16 dummy; (void)dummy;
    {
      // loads + WMMAs; explicit dscnt flush sits between the loads and the barrier
      Frag16 af[2], bg[4];
      #pragma unroll
      for (int mt = 0; mt < 2; ++mt) {
        int row = wm * 32 + mt * 16 + l;
        af[mt].h[0] = *(const v8bf*)(Ab + row * SA + hf * 8);
        af[mt].h[1] = *(const v8bf*)(Ab + row * SA + 16 + hf * 8);
      }
      #pragma unroll
      for (int nt2 = 0; nt2 < 4; ++nt2) {
        int col = wn * 64 + nt2 * 16 + l;
        bg[nt2].h[0] = *(const v8bf*)(Bb + col * SBS + hf * 8);
        bg[nt2].h[1] = *(const v8bf*)(Bb + col * SBS + 16 + hf * 8);
      }
      asm volatile("s_wait_dscnt 0x0" ::: "memory");
      __syncthreads();
      if (t + 2 < nt) issue(t + 2, t & 1);
      #pragma unroll
      for (int mt = 0; mt < 2; ++mt)
        #pragma unroll
        for (int nt2 = 0; nt2 < 4; ++nt2)
          acc[mt][nt2] = __builtin_amdgcn_wmma_f32_16x16x32_bf16(
              false, af[mt].v, false, bg[nt2].v, (short)0, acc[mt][nt2], false, false);
    }
  }
  // peeled final K-step: wait for everything, no further buffer reuse
  asm volatile("s_wait_asynccnt 0x0" ::: "memory");
  __syncthreads();
  wmma_kstep(&As[(nt - 1) & 1][0], &Bs[(nt - 1) & 1][0], acc, wm, wn, l, hf);

  gemm_epilogue<MODE>(acc, m0, n0, wm, wn, l, hf, outF, outB, ldc,
                      bias, res, nullptr, nullptr, nullptr, scale);
}

// ---------------- transpose-staged WMMA GEMM (for reduction-over-token GEMMs) ----------------
// TA=1: A memory [K][M]; TB=1: B memory [K][N]; staged into LDS as [m][k] / [n][k].
template<int TA, int TB, int MODE>
__global__ __launch_bounds__(256) void gemm_bf16(
    const bf16* __restrict__ A, const bf16* __restrict__ B,
    int M, int N, int K, int lda, int ldb,
    float* __restrict__ outF, bf16* __restrict__ outB, int ldc,
    const float* __restrict__ bias, const float* __restrict__ res,
    const float* __restrict__ aux1, const float* __restrict__ aux2,
    const bf16* __restrict__ maskp, float scale)
{
  __shared__ __align__(16) bf16 As[BM * SA];
  __shared__ __align__(16) bf16 Bs[BN * SBS];
  const int tid  = threadIdx.x;
  const int lane = tid & 31, wid = tid >> 5;
  const int wm = wid & 3, wn = wid >> 2;
  const int l = lane & 15, hf = lane >> 4;
  const int m0 = blockIdx.y * BM, n0 = blockIdx.x * BN;

  v8f acc[2][4];
  #pragma unroll
  for (int i = 0; i < 2; ++i)
    #pragma unroll
    for (int j = 0; j < 4; ++j)
      #pragma unroll
      for (int e = 0; e < 8; ++e) acc[i][j][e] = 0.0f;

  for (int k0 = 0; k0 < K; k0 += BKK) {
    if (TA == 0) {
      #pragma unroll
      for (int it = 0; it < 2; ++it) {
        int idx = tid + it * 256;
        int r = idx >> 2, q = idx & 3;
        v8bf vv = *(const v8bf*)(A + (size_t)(m0 + r) * lda + k0 + q * 8);
        *(v8bf*)(&As[r * SA + q * 8]) = vv;
      }
    } else {
      #pragma unroll
      for (int it = 0; it < 2; ++it) {
        int idx = tid + it * 256;
        int kk = idx >> 4, c = idx & 15;
        v8bf vv = *(const v8bf*)(A + (size_t)(k0 + kk) * lda + m0 + c * 8);
        #pragma unroll
        for (int j = 0; j < 8; ++j) As[(c * 8 + j) * SA + kk] = vv[j];
      }
    }
    if (TB == 0) {
      #pragma unroll
      for (int it = 0; it < 2; ++it) {
        int idx = tid + it * 256;
        int r = idx >> 2, q = idx & 3;
        v8bf vv = *(const v8bf*)(B + (size_t)(n0 + r) * ldb + k0 + q * 8);
        *(v8bf*)(&Bs[r * SBS + q * 8]) = vv;
      }
    } else {
      #pragma unroll
      for (int it = 0; it < 2; ++it) {
        int idx = tid + it * 256;
        int kk = idx >> 4, c = idx & 15;
        v8bf vv = *(const v8bf*)(B + (size_t)(k0 + kk) * ldb + n0 + c * 8);
        #pragma unroll
        for (int j = 0; j < 8; ++j) Bs[(c * 8 + j) * SBS + kk] = vv[j];
      }
    }
    __syncthreads();
    wmma_kstep(As, Bs, acc, wm, wn, l, hf);
    __syncthreads();
  }

  gemm_epilogue<MODE>(acc, m0, n0, wm, wn, l, hf, outF, outB, ldc,
                      bias, res, aux1, aux2, maskp, scale);
}

// ---------------- per-head attention to persistent vectors (VALU; 0.4% of FLOPs) ----------------
__global__ __launch_bounds__(256) void attn_kernel(const float* __restrict__ q,
                                                   const float* __restrict__ pv,
                                                   bf16* __restrict__ attn) {
  __shared__ float kvs[PVN][HDIM + 1];
  __shared__ float qs[8][HDIM];
  __shared__ float ps[8][PVN];
  const int h = blockIdx.x;
  for (int idx = threadIdx.x; idx < PVN * HDIM; idx += 256) {
    int p = idx >> 7, d = idx & 127;
    kvs[p][d] = pv[(size_t)p * DDIM + h * HDIM + d];
  }
  const int wid = threadIdx.x >> 5, lane = threadIdx.x & 31;
  const int tok = blockIdx.y * 8 + wid;
  const float* qr = q + (size_t)tok * DDIM + h * HDIM;
  for (int d = lane; d < HDIM; d += 32) qs[wid][d] = qr[d];
  __syncthreads();

  float sc0 = 0.f, sc1 = 0.f;
  for (int d = 0; d < HDIM; ++d) {
    float qv = qs[wid][d];
    sc0 += qv * kvs[lane][d];
    sc1 += qv * kvs[lane + 32][d];
  }
  const float inv = 0.08838834764831845f;  // 1/sqrt(128)
  sc0 *= inv; sc1 *= inv;
  float mx = fmaxf(sc0, sc1);
  for (int o = 16; o > 0; o >>= 1) mx = fmaxf(mx, __shfl_xor(mx, o, 32));
  float e0 = __expf(sc0 - mx), e1 = __expf(sc1 - mx);
  float sum = e0 + e1;
  for (int o = 16; o > 0; o >>= 1) sum += __shfl_xor(sum, o, 32);
  float rinv = 1.0f / sum;
  ps[wid][lane] = e0 * rinv; ps[wid][lane + 32] = e1 * rinv;
  __syncthreads();

  float o0 = 0.f, o1 = 0.f, o2 = 0.f, o3 = 0.f;
  for (int p = 0; p < PVN; ++p) {
    float w = ps[wid][p];
    o0 += w * kvs[p][lane];
    o1 += w * kvs[p][lane + 32];
    o2 += w * kvs[p][lane + 64];
    o3 += w * kvs[p][lane + 96];
  }
  bf16* ar = attn + (size_t)tok * DDIM + h * HDIM;
  ar[lane] = f2bf(o0); ar[lane + 32] = f2bf(o1);
  ar[lane + 64] = f2bf(o2); ar[lane + 96] = f2bf(o3);
}

// ---------------- bias gradient + update: nb = .99*b + .9*mb - .01*sum_rows(mat) ----------------
__global__ __launch_bounds__(256) void bias_update_kernel(const bf16* __restrict__ mat,
                                                          int rows, int C,
                                                          const float* __restrict__ b,
                                                          const float* __restrict__ mb,
                                                          float* __restrict__ nb) {
  int c = blockIdx.x * 256 + threadIdx.x;
  float s = 0.f;
  for (int r = 0; r < rows; ++r) s += bf2f(mat[(size_t)r * C + c]);
  nb[c] = 0.99f * b[c] + 0.9f * mb[c] - 0.01f * s;
}

// ---------------- driver ----------------
extern "C" void kernel_launch(void* const* d_in, const int* in_sizes, int n_in,
                              void* d_out, int out_size, void* d_ws, size_t ws_size,
                              hipStream_t stream) {
  (void)in_sizes; (void)n_in; (void)out_size; (void)ws_size;
  const float* hidden = (const float*)d_in[0];
  const float* pv     = (const float*)d_in[1];
  const float* lng    = (const float*)d_in[2];
  const float* lnb    = (const float*)d_in[3];
  const float* wq  = (const float*)d_in[4];  const float* bq  = (const float*)d_in[5];
  const float* wo  = (const float*)d_in[6];  const float* bo  = (const float*)d_in[7];
  const float* w1  = (const float*)d_in[8];  const float* b1  = (const float*)d_in[9];
  const float* w2  = (const float*)d_in[10]; const float* b2  = (const float*)d_in[11];
  const float* mw1 = (const float*)d_in[12]; const float* mb1 = (const float*)d_in[13];
  const float* mw2 = (const float*)d_in[14]; const float* mb2 = (const float*)d_in[15];
  float* outp = (float*)d_out;

  char* p = (char*)d_ws;
  auto alloc = [&](size_t bytes) -> char* {
    char* r = p; p += (bytes + 255) & ~(size_t)255; return r;
  };
  bf16*  wq16 = (bf16*)alloc((size_t)DDIM * DDIM * 2);
  bf16*  wo16 = (bf16*)alloc((size_t)DDIM * DDIM * 2);
  bf16*  w116 = (bf16*)alloc((size_t)IDIM * DDIM * 2);
  bf16*  w216 = (bf16*)alloc((size_t)DDIM * IDIM * 2);
  bf16*  nw1  = (bf16*)alloc((size_t)IDIM * DDIM * 2);
  bf16*  nw2  = (bf16*)alloc((size_t)DDIM * IDIM * 2);
  float* nb1  = (float*)alloc((size_t)IDIM * 4);
  float* nb2  = (float*)alloc((size_t)DDIM * 4);
  bf16*  hsn  = (bf16*)alloc((size_t)NT * DDIM * 2);   // reused as G after G1
  float* qf   = (float*)alloc((size_t)NT * DDIM * 4);  // reused as dHm (NT*IDIM bf16, same bytes)
  bf16*  at16 = (bf16*)alloc((size_t)NT * DDIM * 2);
  float* pmf  = (float*)alloc((size_t)NT * DDIM * 4);
  bf16*  pm16 = (bf16*)alloc((size_t)NT * DDIM * 2);
  bf16*  Hb   = (bf16*)alloc((size_t)NT * IDIM * 2);   // reused as H2 after grads
  bf16*  Gb   = hsn;
  bf16*  dHm  = (bf16*)qf;

  const float gscale = 2.0f / ((float)NT * (float)DDIM);

  // 0) weights -> bf16
  f2bf_kernel<<<512, 256, 0, stream>>>(wq, wq16, (size_t)DDIM * DDIM);
  f2bf_kernel<<<512, 256, 0, stream>>>(wo, wo16, (size_t)DDIM * DDIM);
  f2bf_kernel<<<512, 256, 0, stream>>>(w1, w116, (size_t)IDIM * DDIM);
  f2bf_kernel<<<512, 256, 0, stream>>>(w2, w216, (size_t)DDIM * IDIM);
  // 1) layernorm
  ln_kernel<<<NT, 256, 0, stream>>>(hidden, lng, lnb, hsn);
  // 2) q = LN(h) @ wq^T + bq
  gemm_bf16_async<0><<<dim3(DDIM/BN, NT/BM), 256, 0, stream>>>(
      hsn, wq16, NT, DDIM, DDIM, DDIM, DDIM, qf, nullptr, DDIM,
      bq, nullptr, 0.f);
  // 3) attention over persistent vectors
  attn_kernel<<<dim3(NH, NT/8), 256, 0, stream>>>(qf, pv, at16);
  // 4) pm = hidden + attn @ wo^T + bo
  gemm_bf16_async<1><<<dim3(DDIM/BN, NT/BM), 256, 0, stream>>>(
      at16, wo16, NT, DDIM, DDIM, DDIM, DDIM, pmf, pm16, DDIM,
      bo, hidden, 0.f);
  // 5) H = relu(pm @ w1^T + b1)
  gemm_bf16_async<2><<<dim3(IDIM/BN, NT/BM), 256, 0, stream>>>(
      pm16, w116, NT, IDIM, DDIM, DDIM, DDIM, nullptr, Hb, IDIM,
      b1, nullptr, 0.f);
  // 6) G = 2/(N*D) * (H @ w2^T + b2 - hidden)
  gemm_bf16_async<3><<<dim3(DDIM/BN, NT/BM), 256, 0, stream>>>(
      Hb, w216, NT, DDIM, IDIM, IDIM, IDIM, nullptr, Gb, DDIM,
      b2, hidden, gscale);
  // 7) new_b2
  bias_update_kernel<<<DDIM/256, 256, 0, stream>>>(Gb, NT, DDIM, b2, mb2, nb2);
  // 8) dHm = relu'(H) * (G @ w2)   [B = w2 as [K=D][N=I], transpose-staged]
  gemm_bf16<0,1,4><<<dim3(IDIM/BN, NT/BM), 256, 0, stream>>>(
      Gb, w216, NT, IDIM, DDIM, DDIM, IDIM, nullptr, dHm, IDIM,
      nullptr, nullptr, nullptr, nullptr, Hb, 0.f);
  // 9) new_b1
  bias_update_kernel<<<IDIM/256, 256, 0, stream>>>(dHm, NT, IDIM, b1, mb1, nb1);
  // 10) new_w2 = .99*w2 + .9*m_w2 - .01 * (G^T @ H)   [M=D, N=I, K=tokens]
  gemm_bf16<1,1,5><<<dim3(IDIM/BN, DDIM/BM), 256, 0, stream>>>(
      Gb, Hb, DDIM, IDIM, NT, DDIM, IDIM, nullptr, nw2, IDIM,
      nullptr, nullptr, w2, mw2, nullptr, 0.f);
  // 11) new_w1 = .99*w1 + .9*m_w1 - .01 * (dHm^T @ pm) [M=I, N=D, K=tokens]
  gemm_bf16<1,1,5><<<dim3(DDIM/BN, IDIM/BM), 256, 0, stream>>>(
      dHm, pm16, IDIM, DDIM, NT, IDIM, DDIM, nullptr, nw1, DDIM,
      nullptr, nullptr, w1, mw1, nullptr, 0.f);
  // 12) H2 = relu(pm @ new_w1^T + new_b1)  (overwrites Hb)
  gemm_bf16_async<2><<<dim3(IDIM/BN, NT/BM), 256, 0, stream>>>(
      pm16, nw1, NT, IDIM, DDIM, DDIM, DDIM, nullptr, Hb, IDIM,
      nb1, nullptr, 0.f);
  // 13) out = pm + H2 @ new_w2^T + new_b2
  gemm_bf16_async<6><<<dim3(DDIM/BN, NT/BM), 256, 0, stream>>>(
      Hb, nw2, NT, DDIM, IDIM, IDIM, IDIM, outp, nullptr, DDIM,
      nb2, pmf, 0.f);
}
